// GraphAttentionLayer_59399397704385
// MI455X (gfx1250) — matrix operand order repaired and checked by reading, not compiled, not run on hardware
//
#include <hip/hip_runtime.h>

// ---------------------------------------------------------------------------
// GAT layer for MI455X (gfx1250, wave32, WMMA bf16 16x16x32)
//   K0 gat_prep : zero score buffers, convert W -> chunk-transposed bf16
//   K1 gat_gemm1: Wh = h@W (bf16 WMMA, pipelined), emit Whb_t + s1/s2 scores
//   K2 gat_attn : fused masked-softmax + attn@Wh, 64x64 block (4 waves x
//                 16 rows), double-buffered adj/B loads, adj streamed once
//                 with NT hints (134 MB -> memory-roof bound)
// ---------------------------------------------------------------------------

#define ALPHA 0.2f

typedef __attribute__((ext_vector_type(16))) __bf16 v16bf;
typedef __attribute__((ext_vector_type(8)))  float  v8f;
typedef __attribute__((ext_vector_type(4)))  int    i32x4;   // clang vector: OK for
                                                             // __builtin_nontemporal_load

union ABFrag { unsigned int u[8]; v16bf v; };          // A fragment (16x32 bf16)
union BLoad  { uint4 q[2]; v16bf v; };                 // B fragment (two 16B loads: k 8h..8h+7, 16+8h..16+8h+7)

// round-to-nearest-even single conversion (cold paths)
__device__ __forceinline__ unsigned short f32_to_bf16(float f) {
    union { float f; unsigned int u; } c; c.f = f;
    unsigned int u = c.u;
    u += 0x7FFFu + ((u >> 16) & 1u);
    return (unsigned short)(u >> 16);
}
// fast pack for the hot loop: +0x8000 round, then one v_perm_b32
__device__ __forceinline__ unsigned int pack_rb(float lo, float hi) {
    union { float f; unsigned int u; } a, b; a.f = lo; b.f = hi;
    return __builtin_amdgcn_perm(b.u + 0x8000u, a.u + 0x8000u, 0x07060302u);
}
__device__ __forceinline__ v8f v8f_zero() {
    v8f r;
#pragma unroll
    for (int i = 0; i < 8; ++i) r[i] = 0.0f;
    return r;
}

__device__ __forceinline__ void wmma4(v8f acc[4], const ABFrag& af, const BLoad bt[4]) {
#pragma unroll
    for (int t = 0; t < 4; ++t)
        acc[t] = __builtin_amdgcn_wmma_f32_16x16x32_bf16(
            false, af.v, false, bt[t].v, (short)0, acc[t], false, false);
}

// ---------------------------------------------------------------------------
// K0: zero s1/s2 accumulators; W[256][256] f32 -> Wbt[k/32][n][k%32] bf16
// ---------------------------------------------------------------------------
__global__ void gat_prep(const float* __restrict__ W,
                         unsigned short* __restrict__ Wbt,
                         float* __restrict__ s1g, float* __restrict__ s2g) {
    int idx = blockIdx.x * 256 + threadIdx.x;          // 0..65535
    if (idx < 16384) { s1g[idx] = 0.0f; s2g[idx] = 0.0f; }
    int k = idx >> 8, n = idx & 255;
    Wbt[(((k >> 5) * 256 + n) << 5) + (k & 31)] = f32_to_bf16(W[idx]);
}

// ---------------------------------------------------------------------------
// K1: Wh = h @ W. Block = 64 rows x 64 cols, wave w owns rows r0+16w.
// ---------------------------------------------------------------------------
struct PFG { float4 hq[4]; BLoad bt[4]; };

__device__ __forceinline__ void g1_prefetch(PFG& p, const float* hrow, int k0, int koff,
                                            const unsigned short* bchunk) {
    p.hq[0] = *(const float4*)(hrow + k0 + koff);
    p.hq[1] = *(const float4*)(hrow + k0 + koff + 4);
    p.hq[2] = *(const float4*)(hrow + k0 + koff + 16);
    p.hq[3] = *(const float4*)(hrow + k0 + koff + 20);
#pragma unroll
    for (int t = 0; t < 4; ++t) {
        const unsigned short* q = bchunk + (t << 9);
        p.bt[t].q[0] = *(const uint4*)(q);
        p.bt[t].q[1] = *(const uint4*)(q + 16);
    }
}

__global__ __launch_bounds__(128) void gat_gemm1(
        const float* __restrict__ h, const unsigned short* __restrict__ Wbt,
        const float* __restrict__ a_vec,
        float* __restrict__ s1g, float* __restrict__ s2g,
        unsigned short* __restrict__ Whbt) {
    const int N = 2048, F = 256;
    int blk    = blockIdx.x;
    int colblk = blk & 3;
    int rowblk = (blk >> 2) & 31;
    int b      = blk >> 7;
    int tid  = threadIdx.x;
    int wave = tid >> 5;
    int lane = tid & 31;
    int l16  = lane & 15;
    int half = lane >> 4;
    int r0   = (rowblk << 6) + (wave << 4);
    int col0 = colblk << 6;
    int koff = half << 3;

    v8f acc[4];
#pragma unroll
    for (int t = 0; t < 4; ++t) acc[t] = v8f_zero();

    const float* hrow = h + ((size_t)b * N + (r0 + l16)) * F;
    const unsigned short* bbase = Wbt + ((size_t)(col0 + l16) << 5) + (half << 3);

    PFG p0, p1;
    g1_prefetch(p0, hrow, 0, koff, bbase);
#pragma unroll 1
    for (int kc = 0; kc < 8; kc += 2) {
        g1_prefetch(p1, hrow, (kc + 1) << 5, koff, bbase + (size_t)(kc + 1) * 8192);
        {
            ABFrag af;
            af.u[0] = pack_rb(p0.hq[0].x, p0.hq[0].y);
            af.u[1] = pack_rb(p0.hq[0].z, p0.hq[0].w);
            af.u[2] = pack_rb(p0.hq[1].x, p0.hq[1].y);
            af.u[3] = pack_rb(p0.hq[1].z, p0.hq[1].w);
            af.u[4] = pack_rb(p0.hq[2].x, p0.hq[2].y);
            af.u[5] = pack_rb(p0.hq[2].z, p0.hq[2].w);
            af.u[6] = pack_rb(p0.hq[3].x, p0.hq[3].y);
            af.u[7] = pack_rb(p0.hq[3].z, p0.hq[3].w);
            wmma4(acc, af, p0.bt);
        }
        if (kc + 2 < 8)
            g1_prefetch(p0, hrow, (kc + 2) << 5, koff, bbase + (size_t)(kc + 2) * 8192);
        {
            ABFrag af;
            af.u[0] = pack_rb(p1.hq[0].x, p1.hq[0].y);
            af.u[1] = pack_rb(p1.hq[0].z, p1.hq[0].w);
            af.u[2] = pack_rb(p1.hq[1].x, p1.hq[1].y);
            af.u[3] = pack_rb(p1.hq[1].z, p1.hq[1].w);
            af.u[4] = pack_rb(p1.hq[2].x, p1.hq[2].y);
            af.u[5] = pack_rb(p1.hq[2].z, p1.hq[2].w);
            af.u[6] = pack_rb(p1.hq[3].x, p1.hq[3].y);
            af.u[7] = pack_rb(p1.hq[3].z, p1.hq[3].w);
            wmma4(acc, af, p1.bt);
        }
    }

    // ---- epilogue: Whb_t writes + score partials ----
    float pr1[8], pr2[8];
#pragma unroll
    for (int r = 0; r < 8; ++r) { pr1[r] = 0.0f; pr2[r] = 0.0f; }

#pragma unroll
    for (int t = 0; t < 4; ++t) {
        int col = col0 + (t << 4) + l16;
        float a1 = a_vec[col];
        float a2 = a_vec[256 + col];
#pragma unroll
        for (int r = 0; r < 8; ++r) {
            float v = acc[t][r];
            int n  = r0 + r + (half << 3);
            int jc = n >> 5, jj = n & 31;
            Whbt[((((size_t)b << 6) + jc) * 256 + col) * 32 + jj] = f32_to_bf16(v);
            pr1[r] += v * a1;
            pr2[r] += v * a2;
        }
    }
#pragma unroll
    for (int r = 0; r < 8; ++r) {
        float v1 = pr1[r], v2 = pr2[r];
#pragma unroll
        for (int m = 1; m < 16; m <<= 1) {             // reduce within 16-lane half
            v1 += __shfl_xor(v1, m, 32);
            v2 += __shfl_xor(v2, m, 32);
        }
        if (l16 == 0) {
            int n = r0 + r + (half << 3);
            atomicAdd(&s1g[b * 2048 + n], v1);
            atomicAdd(&s2g[b * 2048 + n], v2);
        }
    }
}

// ---------------------------------------------------------------------------
// K2: fused masked softmax + attn @ Wh.
//     p = adj ? exp(lrelu(s1_i+s2_j) - M_i) : 0 with M_i = lrelu(s1_i+max s2)
//     Block = 64 rows x 64 cols; wave w owns rows r0+16w (no redundant exp;
//     B fragments shared across waves through WGP$). Double-buffered loads.
// ---------------------------------------------------------------------------
struct PFA { i32x4 ad[4]; BLoad bt[4]; };

__device__ __forceinline__ void at_prefetch(PFA& p, const int* adjrow, int j0, int koff,
                                            const unsigned short* bchunk) {
    p.ad[0] = __builtin_nontemporal_load((const i32x4*)(adjrow + j0 + koff));
    p.ad[1] = __builtin_nontemporal_load((const i32x4*)(adjrow + j0 + koff + 4));
    p.ad[2] = __builtin_nontemporal_load((const i32x4*)(adjrow + j0 + koff + 16));
    p.ad[3] = __builtin_nontemporal_load((const i32x4*)(adjrow + j0 + koff + 20));
#pragma unroll
    for (int t = 0; t < 4; ++t) {
        const unsigned short* q = bchunk + (t << 9);
        p.bt[t].q[0] = *(const uint4*)(q);
        p.bt[t].q[1] = *(const uint4*)(q + 16);
    }
}

__device__ __forceinline__ void at_buildA(ABFrag& af, const PFA& p, const float* s2s,
                                          int j0, int koff, float s1v, float mb,
                                          float& lsum) {
    float4 sa = *(const float4*)(s2s + j0 + koff);
    float4 sb = *(const float4*)(s2s + j0 + koff + 4);
    float4 sc = *(const float4*)(s2s + j0 + koff + 16);
    float4 sd = *(const float4*)(s2s + j0 + koff + 20);
#define GAT_PAIR(vv, a0, a1, x0, x1)                                  \
    {                                                                 \
        float e0 = s1v + (x0); e0 = (e0 > 0.0f) ? e0 : ALPHA * e0;    \
        float e1 = s1v + (x1); e1 = (e1 > 0.0f) ? e1 : ALPHA * e1;    \
        float q0 = ((a0) > 0) ? __expf(e0 - mb) : 0.0f;               \
        float q1 = ((a1) > 0) ? __expf(e1 - mb) : 0.0f;               \
        lsum += q0 + q1;                                              \
        af.u[vv] = pack_rb(q0, q1);                                   \
    }
    GAT_PAIR(0, p.ad[0].x, p.ad[0].y, sa.x, sa.y)
    GAT_PAIR(1, p.ad[0].z, p.ad[0].w, sa.z, sa.w)
    GAT_PAIR(2, p.ad[1].x, p.ad[1].y, sb.x, sb.y)
    GAT_PAIR(3, p.ad[1].z, p.ad[1].w, sb.z, sb.w)
    GAT_PAIR(4, p.ad[2].x, p.ad[2].y, sc.x, sc.y)
    GAT_PAIR(5, p.ad[2].z, p.ad[2].w, sc.z, sc.w)
    GAT_PAIR(6, p.ad[3].x, p.ad[3].y, sd.x, sd.y)
    GAT_PAIR(7, p.ad[3].z, p.ad[3].w, sd.z, sd.w)
#undef GAT_PAIR
}

__global__ __launch_bounds__(128) void gat_attn(
        const int* __restrict__ adj, const float* __restrict__ s1g,
        const float* __restrict__ s2g, const unsigned short* __restrict__ Whbt,
        float* __restrict__ out) {
    const int N = 2048, F = 256;
    __shared__ float s2s[2048];
    __shared__ float red[128];
    __shared__ float s2max_s;
    __shared__ float linv_s[4][16];

    int blk    = blockIdx.x;
    int colblk = blk & 3;                               // fastest-varying: the 4
    int rowblk = (blk >> 2) & 31;                       // col-blocks sharing adj
    int b      = blk >> 7;                              // rows run adjacently
    int tid  = threadIdx.x;
    int wave = tid >> 5;
    int lane = tid & 31;
    int l16  = lane & 15;
    int half = lane >> 4;
    int r0   = (rowblk << 6) + (wave << 4);
    int col0 = colblk << 6;
    int koff = half << 3;

    // stage s2 row in LDS + batch max
    float lm = -3.0e38f;
    for (int j = tid; j < N; j += 128) {
        float v = s2g[b * 2048 + j];
        s2s[j] = v;
        lm = fmaxf(lm, v);
    }
    red[tid] = lm;
    __syncthreads();
    if (tid == 0) {
        float m = red[0];
        for (int i = 1; i < 128; ++i) m = fmaxf(m, red[i]);
        s2max_s = m;
    }
    __syncthreads();
    float s2max = s2max_s;

    int   myrow = r0 + l16;
    float s1v   = s1g[b * 2048 + myrow];
    float mb    = s1v + s2max;
    mb = (mb > 0.0f) ? mb : ALPHA * mb;                 // row-wise logit bound

    const int* adjrow = adj + ((size_t)b * N + myrow) * N;
    const unsigned short* bbase =
        Whbt + (((size_t)b << 6) * 256 + col0 + l16) * 32 + (half << 3);

    v8f acc[4];
#pragma unroll
    for (int t = 0; t < 4; ++t) acc[t] = v8f_zero();
    float lsum = 0.0f;

    PFA p0, p1;
    at_prefetch(p0, adjrow, 0, koff, bbase);
#pragma unroll 1
    for (int jc = 0; jc < 64; jc += 2) {
        at_prefetch(p1, adjrow, (jc + 1) << 5, koff, bbase + (size_t)(jc + 1) * 8192);
        {
            ABFrag af;
            at_buildA(af, p0, s2s, jc << 5, koff, s1v, mb, lsum);
            wmma4(acc, af, p0.bt);
        }
        if (jc + 2 < 64)
            at_prefetch(p0, adjrow, (jc + 2) << 5, koff, bbase + (size_t)(jc + 2) * 8192);
        {
            ABFrag af;
            at_buildA(af, p1, s2s, (jc + 1) << 5, koff, s1v, mb, lsum);
            wmma4(acc, af, p1.bt);
        }
    }

    // row sums: lane l and l+16 hold disjoint j halves of row l16
    lsum += __shfl_xor(lsum, 16, 32);
    if (half == 0) linv_s[wave][l16] = 1.0f / fmaxf(lsum, 1e-30f);
    __syncthreads();

    float li[8];
#pragma unroll
    for (int r = 0; r < 8; ++r) li[r] = linv_s[wave][r + (half << 3)];

#pragma unroll
    for (int t = 0; t < 4; ++t) {
        int col = col0 + (t << 4) + l16;
#pragma unroll
        for (int r = 0; r < 8; ++r) {
            int n = r0 + r + (half << 3);
            __builtin_nontemporal_store(acc[t][r] * li[r],
                                        &out[((size_t)b * N + n) * F + col]);
        }
    }
}

// ---------------------------------------------------------------------------
extern "C" void kernel_launch(void* const* d_in, const int* in_sizes, int n_in,
                              void* d_out, int out_size, void* d_ws, size_t ws_size,
                              hipStream_t stream) {
    const float* h   = (const float*)d_in[0];   // (8,2048,256) f32
    const int*   adj = (const int*)  d_in[1];   // (8,2048,2048) i32
    const float* W   = (const float*)d_in[2];   // (256,256) f32
    const float* a   = (const float*)d_in[3];   // (512,1) f32
    float*       out = (float*)d_out;           // (8,2048,256) f32

    // workspace: s1 (64KB) | s2 (64KB) | Wbt (128KB) | Whb_t (8MB)
    float*          s1g  = (float*)d_ws;
    float*          s2g  = s1g + 16384;
    unsigned short* Wbt  = (unsigned short*)(s2g + 16384);
    unsigned short* Whbt = Wbt + 65536;

    hipLaunchKernelGGL(gat_prep,  dim3(256),  dim3(256), 0, stream, W, Wbt, s1g, s2g);
    hipLaunchKernelGGL(gat_gemm1, dim3(1024), dim3(128), 0, stream, h, Wbt, a, s1g, s2g, Whbt);
    hipLaunchKernelGGL(gat_attn,  dim3(1024), dim3(128), 0, stream, adj, s1g, s2g, Whbt, out);
}
